// ClassTemporalAlignmentBlock_14869176779000
// MI455X (gfx1250) — compile-verified, matrix operand order checked by reading, not run on hardware
//
#include <hip/hip_runtime.h>
#include <hip/hip_bf16.h>

typedef __bf16 bf16;
typedef bf16  bf16x16 __attribute__((ext_vector_type(16)));
typedef float f32x8   __attribute__((ext_vector_type(8)));

#define B_   32
#define H_   48
#define W_   48
#define C_   128
#define HID  256
#define T_   (H_ * W_)          // 2304
#define M_   (B_ * T_)          // 73728
#define G3   (3 * HID)          // 768
#define HSL  128                // hidden slice per workgroup in the recurrence
#define NWGD 2                  // workgroups per direction

__device__ __forceinline__ f32x8 wmma_bf16(bf16x16 a, bf16x16 b, f32x8 c) {
  // D = A(16x32) * B(32x16) + C(16x16), f32 accumulate
  return __builtin_amdgcn_wmma_f32_16x16x32_bf16(
      /*neg_a=*/false, a, /*neg_b=*/false, b,
      /*c_mod=*/(short)0, c, /*reuse_a=*/false, /*reuse_b=*/false);
}

// Async global->LDS copy (CDNA5): 16B per issue, tracked by ASYNCcnt.
// VDST = VGPR holding LDS byte address; VADDR = 64-bit global address; SADDR=off.
__device__ __forceinline__ void async_g2l_b128(unsigned lds_byte_addr,
                                               const void* gaddr) {
  asm volatile("global_load_async_to_lds_b128 %0, %1, off"
               :: "v"(lds_byte_addr), "v"(gaddr)
               : "memory");
}
__device__ __forceinline__ void wait_async0() {
  asm volatile("s_wait_asynccnt 0" ::: "memory");
}
// Truncate generic pointer to LDS byte address (low 32 bits).
__device__ __forceinline__ unsigned lds_addr(const void* p) {
  return (unsigned)(unsigned long long)p;
}

// ---------------------------------------------------------------------------
// Conversion kernels: f32 -> bf16, and f32 [K][N] -> bf16 [N][K] (transpose)
// ---------------------------------------------------------------------------
__global__ void k_cvt_bf16(const float* __restrict__ src, bf16* __restrict__ dst, long n) {
  long i = (long)blockIdx.x * blockDim.x + threadIdx.x;
  if (i < n) dst[i] = (bf16)src[i];
}

__global__ void k_cvt_transpose(const float* __restrict__ src, bf16* __restrict__ dst,
                                int K, int N) {
  long i = (long)blockIdx.x * blockDim.x + threadIdx.x;
  if (i < (long)K * N) {
    int k = (int)(i / N), n = (int)(i % N);
    dst[(long)n * K + k] = (bf16)src[i];
  }
}

__global__ void k_init_barriers(unsigned int* ctr) {
  if (threadIdx.x < 8) ctr[threadIdx.x] = 0u;
}

// ---------------------------------------------------------------------------
// Generic bf16 WMMA GEMM:  C[M x N] = A[M x K] * B^T[N x K] (+ bias[N])
// One 16x16 output tile per wave, K-loop in steps of 32.
// ---------------------------------------------------------------------------
template <typename OUT_T>
__global__ __launch_bounds__(256)
void k_gemm_bf16(const bf16* __restrict__ A, const bf16* __restrict__ BT,
                 const float* __restrict__ bias, OUT_T* __restrict__ C,
                 int M, int N, int K) {
  const int wave = threadIdx.x >> 5;
  const unsigned lane = threadIdx.x & 31;
  const int ntn = N >> 4;
  long tile = (long)blockIdx.x * 8 + wave;
  long mt = tile / ntn;
  int  nt = (int)(tile % ntn);
  int m0 = (int)mt * 16, n0 = nt * 16;
  if (m0 >= M) return;

  f32x8 acc = {};
  const bf16* ap = A  + (long)(m0 + (lane & 15)) * K + 16 * (lane >> 4);
  const bf16* bp = BT + (long)(n0 + (lane & 15)) * K + 16 * (lane >> 4);
#pragma unroll 4
  for (int k = 0; k < K; k += 32) {
    bf16x16 a = *(const bf16x16*)(ap + k);
    bf16x16 b = *(const bf16x16*)(bp + k);
    acc = wmma_bf16(a, b, acc);
  }
  const int col  = n0 + (lane & 15);
  const float bv = bias ? bias[col] : 0.0f;
  const int row0 = m0 + (int)(lane >> 4) * 8;
#pragma unroll
  for (int i = 0; i < 8; ++i) {
    float v = acc[i] + bv;
    C[(long)(row0 + i) * N + col] = (OUT_T)v;
  }
}

// ---------------------------------------------------------------------------
// Bidirectional GRU recurrence.
// Grid = 4 workgroups: blockIdx = {dir, half}. Each WG owns HSL=128 hidden
// units (z/r/h gate columns, 256x384 bf16 = 192 KB LDS-resident), computes
// inner = h @ rkernel_slice via WMMA each step, applies gates, exchanges its
// h-half with its sibling WG through L2 + atomic phase barrier. All
// global->LDS staging uses async-to-LDS (ASYNCcnt) loads.
// ---------------------------------------------------------------------------
__global__ __launch_bounds__(256)
void k_bigru(const float* __restrict__ XP,   // [2][B][T][768] f32 (bias0 added)
             const bf16*  __restrict__ RT,   // [2][768][256]  bf16 (rkernel^T)
             const float* __restrict__ fw_bias,  // [2][768]
             const float* __restrict__ bw_bias,  // [2][768]
             bf16* __restrict__ hx,           // [2][B][HID] exchange buffer
             unsigned int* __restrict__ barrier_ctr,  // [2]
             float* __restrict__ out) {       // [B][T][2*HID]
  extern __shared__ unsigned char smem[];
  bf16*  sRT  = (bf16*)smem;                                      // [3*HSL][HID] 192KB
  bf16*  sH   = (bf16*)(smem + (size_t)3 * HSL * HID * 2);        // [B][HID]     16KB
  float* sInn = (float*)(smem + (size_t)3 * HSL * HID * 2
                              + (size_t)B_ * HID * 2);            // [B][3*HSL]   48KB
  float* sHp  = (float*)(smem + (size_t)3 * HSL * HID * 2
                              + (size_t)B_ * HID * 2
                              + (size_t)B_ * 3 * HSL * 4);        // [B][HSL]     16KB

  const int dir  = blockIdx.x >> 1;
  const int half = blockIdx.x & 1;
  const int j0   = half * HSL;
  const int tid  = threadIdx.x;
  const unsigned lane = tid & 31;
  const int wave = tid >> 5;

  const bf16*  rt    = RT + (long)dir * G3 * HID;
  const float* xp    = XP + (long)dir * (long)M_ * G3;
  const float* bias1 = (dir == 0 ? fw_bias : bw_bias) + G3;  // bias row 1
  unsigned int* bar  = barrier_ctr + dir;

  // Fill LDS-resident rkernel slice with async-to-LDS loads:
  // gate g uses rows [g*HID + j0, +HSL) of RT.
  for (int g = 0; g < 3; ++g) {
    const bf16* src = rt + (long)(g * HID + j0) * HID;
    const unsigned dbase = lds_addr(sRT + (long)(g * HSL) * HID);
    const int nvec = HSL * HID / 8;  // 16B chunks
    for (int v = tid; v < nvec; v += 256)
      async_g2l_b128(dbase + (unsigned)v * 16u, src + (long)v * 8);
  }
  // h0 = 0
  {
    uint4 z; z.x = z.y = z.z = z.w = 0u;
    for (int v = tid; v < B_ * HID / 8; v += 256) ((uint4*)sH)[v] = z;
    for (int i = tid; i < B_ * HSL; i += 256) sHp[i] = 0.0f;
  }
  wait_async0();
  __syncthreads();

  for (int t = 0; t < T_; ++t) {
    const int p = (dir == 0) ? t : (T_ - 1 - t);

    // inner = sH[32x256] @ slice^T -> [32 x 384]; wave w owns n-tile w of each gate.
    f32x8 acc[2][3] = {};
    const bf16* aBase = sH + (long)(lane & 15) * HID + 16 * (lane >> 4);
#pragma unroll
    for (int g = 0; g < 3; ++g) {
      const bf16* bBase = sRT + (long)(g * HSL + wave * 16 + (lane & 15)) * HID
                              + 16 * (lane >> 4);
#pragma unroll
      for (int k = 0; k < HID; k += 32) {
        bf16x16 b  = *(const bf16x16*)(bBase + k);
        bf16x16 a0 = *(const bf16x16*)(aBase + k);
        bf16x16 a1 = *(const bf16x16*)(aBase + 16 * HID + k);
        acc[0][g] = wmma_bf16(a0, b, acc[0][g]);
        acc[1][g] = wmma_bf16(a1, b, acc[1][g]);
      }
    }
    // Accumulators -> LDS (add recurrent bias row)
#pragma unroll
    for (int g = 0; g < 3; ++g) {
      const int col  = g * HSL + wave * 16 + (int)(lane & 15);
      const int gcol = g * HID + j0 + wave * 16 + (int)(lane & 15);
      const float bb = bias1[gcol];
#pragma unroll
      for (int mt = 0; mt < 2; ++mt) {
        const int row0 = mt * 16 + (int)(lane >> 4) * 8;
#pragma unroll
        for (int i = 0; i < 8; ++i)
          sInn[(row0 + i) * (3 * HSL) + col] = acc[mt][g][i] + bb;
      }
    }
    __syncthreads();

    // Elementwise gate update for this WG's 32x128 slice.
    for (int e = tid; e < B_ * HSL; e += 256) {
      const int b  = e / HSL, jj = e % HSL;
      const int j  = j0 + jj;
      const float* xrow = xp + ((long)b * T_ + p) * G3;
      if (t + 1 < T_) {
        const int pn = (dir == 0) ? (p + 1) : (p - 1);
        __builtin_prefetch(xp + ((long)b * T_ + pn) * G3 + j, 0, 1);
      }
      const float iz = sInn[b * (3 * HSL) + jj];
      const float ir = sInn[b * (3 * HSL) + HSL + jj];
      const float ih = sInn[b * (3 * HSL) + 2 * HSL + jj];
      const float xz = xrow[j];
      const float xr = xrow[HID + j];
      const float xh = xrow[2 * HID + j];
      const float z  = 1.0f / (1.0f + __expf(-(xz + iz)));
      const float r  = 1.0f / (1.0f + __expf(-(xr + ir)));
      const float hh = tanhf(xh + r * ih);
      const float hn = z * sHp[e] + (1.0f - z) * hh;
      sHp[e] = hn;
      sH[b * HID + j] = (bf16)hn;
      out[((long)b * T_ + p) * (2 * HID) + dir * HID + j] = hn;
      hx[((long)dir * B_ + b) * HID + j] = (bf16)hn;
    }
    __threadfence();
    __syncthreads();

    // 2-member phase barrier with the sibling WG of this direction.
    if (tid == 0) {
      atomicAdd(bar, 1u);
      const unsigned target = (unsigned)(t + 1) * NWGD;
      while (atomicAdd(bar, 0u) < target) __builtin_amdgcn_s_sleep(1);
    }
    __syncthreads();
    __threadfence();

    // Pull sibling's h-half into sH via async-to-LDS (8KB, 2 issues/thread).
    {
      const int oj0 = (1 - half) * HSL;
      const unsigned hbase = lds_addr(sH);
      for (int e = tid; e < B_ * HSL / 8; e += 256) {
        const int b   = e / (HSL / 8);
        const int off = (e % (HSL / 8)) * 8;
        async_g2l_b128(hbase + (unsigned)(b * HID + oj0 + off) * 2u,
                       hx + ((long)dir * B_ + b) * HID + oj0 + off);
      }
      wait_async0();
      __syncthreads();
    }
  }
}

// ---------------------------------------------------------------------------
// Host launch
// ---------------------------------------------------------------------------
static inline size_t align256(size_t x) { return (x + 255) & ~(size_t)255; }

extern "C" void kernel_launch(void* const* d_in, const int* in_sizes, int n_in,
                              void* d_out, int out_size, void* d_ws, size_t ws_size,
                              hipStream_t stream) {
  const float* x       = (const float*)d_in[0];  // [B,H,W,C]
  const float* proj_w  = (const float*)d_in[1];  // [128,256]
  const float* proj_b  = (const float*)d_in[2];  // [256]
  const float* fw_k    = (const float*)d_in[3];  // [256,768]
  const float* fw_rk   = (const float*)d_in[4];  // [256,768]
  const float* fw_bias = (const float*)d_in[5];  // [2,768]
  const float* bw_k    = (const float*)d_in[6];
  const float* bw_rk   = (const float*)d_in[7];
  const float* bw_bias = (const float*)d_in[8];
  float* out = (float*)d_out;

  // Workspace carve-up
  unsigned char* ws = (unsigned char*)d_ws;
  size_t off = 0;
  bf16* xb    = (bf16*)(ws + off); off = align256(off + (size_t)M_ * C_ * 2);
  bf16* projT = (bf16*)(ws + off); off = align256(off + (size_t)HID * C_ * 2);
  bf16* fwKT  = (bf16*)(ws + off); off = align256(off + (size_t)G3 * HID * 2);
  bf16* bwKT  = (bf16*)(ws + off); off = align256(off + (size_t)G3 * HID * 2);
  bf16* RT    = (bf16*)(ws + off); off = align256(off + (size_t)2 * G3 * HID * 2);
  bf16* E     = (bf16*)(ws + off); off = align256(off + (size_t)M_ * HID * 2);
  float* XP   = (float*)(ws + off); off = align256(off + (size_t)2 * M_ * G3 * 4);
  bf16* hx    = (bf16*)(ws + off); off = align256(off + (size_t)2 * B_ * HID * 2);
  unsigned int* bar = (unsigned int*)(ws + off); off = align256(off + 256);
  (void)ws_size; (void)n_in; (void)in_sizes; (void)out_size;

  // 1) Conversions / transposes to bf16
  {
    long n = (long)M_ * C_;
    k_cvt_bf16<<<(unsigned)((n + 255) / 256), 256, 0, stream>>>(x, xb, n);
  }
  {
    long n = (long)C_ * HID;
    k_cvt_transpose<<<(unsigned)((n + 255) / 256), 256, 0, stream>>>(proj_w, projT, C_, HID);
  }
  {
    long n = (long)HID * G3;
    unsigned g = (unsigned)((n + 255) / 256);
    k_cvt_transpose<<<g, 256, 0, stream>>>(fw_k,  fwKT,            HID, G3);
    k_cvt_transpose<<<g, 256, 0, stream>>>(bw_k,  bwKT,            HID, G3);
    k_cvt_transpose<<<g, 256, 0, stream>>>(fw_rk, RT,              HID, G3);
    k_cvt_transpose<<<g, 256, 0, stream>>>(bw_rk, RT + (long)G3 * HID, HID, G3);
  }

  // 2) E = x @ proj_w + proj_b  (bf16 out)
  {
    long tiles = ((long)M_ / 16) * (HID / 16);
    k_gemm_bf16<bf16><<<(unsigned)((tiles + 7) / 8), 256, 0, stream>>>(
        xb, projT, proj_b, E, M_, HID, C_);
  }
  // 3) XP[dir] = E @ kernel + bias[0]  (f32 out)
  {
    long tiles = ((long)M_ / 16) * (G3 / 16);
    unsigned g = (unsigned)((tiles + 7) / 8);
    k_gemm_bf16<float><<<g, 256, 0, stream>>>(E, fwKT, fw_bias, XP, M_, G3, HID);
    k_gemm_bf16<float><<<g, 256, 0, stream>>>(E, bwKT, bw_bias, XP + (long)M_ * G3,
                                              M_, G3, HID);
  }

  // 4) Recurrence
  k_init_barriers<<<1, 32, 0, stream>>>(bar);
  const size_t smem = (size_t)3 * HSL * HID * 2   // sRT 192KB
                    + (size_t)B_ * HID * 2        // sH   16KB
                    + (size_t)B_ * 3 * HSL * 4    // sInn 48KB
                    + (size_t)B_ * HSL * 4;       // sHp  16KB  => 272KB
  hipFuncSetAttribute((const void*)k_bigru,
                      hipFuncAttributeMaxDynamicSharedMemorySize, (int)smem);
  k_bigru<<<4, 256, smem, stream>>>(XP, RT, fw_bias, bw_bias, hx, bar, out);
}
